// AttentionBlock_49761491091725
// MI455X (gfx1250) — compile-verified, matrix operand order checked by reading, not run on hardware
//
#include <hip/hip_runtime.h>

// MI455X (gfx1250, wave32) attention block:
//   GroupNorm -> QKV GEMM -> 4-head flash attention -> proj GEMM + residual.
// All matmuls on v_wmma_f32_16x16x32_f16; fragments are pairs of b128 loads;
// B operands staged transposed in LDS; attention V tiles DMA'd by the
// Tensor Data Mover (tensor_load_to_lds + s_wait_tensorcnt).

typedef __attribute__((ext_vector_type(16))) _Float16 v16h;
typedef __attribute__((ext_vector_type(8)))  _Float16 v8h;
typedef __attribute__((ext_vector_type(4)))  _Float16 v4h;
typedef __attribute__((ext_vector_type(8)))  float    v8f;
typedef __attribute__((ext_vector_type(4)))  unsigned int u32x4;
typedef __attribute__((ext_vector_type(8)))  int      i32x8;
typedef __attribute__((ext_vector_type(4)))  int      i32x4;

#define B_      16
#define C_      256
#define N_      1024
#define HEADS_  4
#define GROUPS_ 32
#define CPG_    8

// amdgpu-toolchain (clang-23/therock-10.0) ships this header and uses the
// 6-arg tensor_load_to_lds builtin; ROCm 7.2 (clang-22) uses the 5-arg form.
#if defined(__has_include)
#if __has_include(<hip/amd_detail/amd_gfx1250_TDM.h>)
#define TDM_ARGS6 1
#else
#define TDM_ARGS6 0
#endif
#else
#define TDM_ARGS6 0
#endif

// ---------------------------------------------------------------------------
// WMMA helpers (ISA 7.12.2 layouts, wave32)
// ---------------------------------------------------------------------------
__device__ __forceinline__ v8f wmma_f16(v16h a, v16h b, v8f c) {
  return __builtin_amdgcn_wmma_f32_16x16x32_f16(false, a, false, b, (short)0, c,
                                                false, false);
}

// A-matrix 16x32 f16 fragment from row-major p (element (m,k) = p[m*ld+k]).
// Per lane the 16 values are two contiguous 8xf16 runs -> two b128 loads.
__device__ __forceinline__ v16h frag_a(const _Float16* __restrict__ p, int ld) {
  const int lane = threadIdx.x & 31;
  const int m = lane & 15, hh = lane >> 4;
  const v8h lo = *(const v8h*)(p + m * ld + hh * 8);
  const v8h hi = *(const v8h*)(p + m * ld + hh * 8 + 16);
  return __builtin_shufflevector(lo, hi, 0, 1, 2, 3, 4, 5, 6, 7, 8, 9, 10, 11,
                                 12, 13, 14, 15);
}
// B fragment == A fragment read from the transposed buffer.

// ---------------------------------------------------------------------------
// Tensor Data Mover: 2D f16 tile (tile_x elems wide, tile_y rows, row stride
// `stride` elems) from global -> LDS at lds_byte_addr (compact rows).
// D# packing per CDNA5 ISA ch.8. Wave-level op; EXEC ignored.
// ---------------------------------------------------------------------------
__device__ __forceinline__ void tdm_load_tile_f16(unsigned lds_byte_addr,
                                                  const _Float16* gptr,
                                                  unsigned tile_x,
                                                  unsigned tile_y,
                                                  unsigned stride) {
  const unsigned long long ga = (unsigned long long)(uintptr_t)gptr;
  u32x4 g0 = {0u, 0u, 0u, 0u};
  g0[0] = 1u;                                   // count=1, user descriptor
  g0[1] = lds_byte_addr;                        // LDS byte address
  g0[2] = (unsigned)ga;                         // global_addr[31:0]
  g0[3] = (unsigned)((ga >> 32) & 0x01ffffffu)  // global_addr[56:32]
          | 0x80000000u;                        // type=2 ("image")
  i32x8 g1 = {0, 0, 0, 0, 0, 0, 0, 0};
  const unsigned td0 = stride;                  // tensor_dim0 (elems)
  const unsigned td1 = tile_y;                  // tensor_dim1
  g1[0] = 0x00010000;                           // data_size=1 -> 2 bytes
  g1[1] = (int)((td0 & 0xffffu) << 16);         // tensor_dim0[15:0]
  g1[2] = (int)((td0 >> 16) | ((td1 & 0xffffu) << 16));
  g1[3] = (int)((td1 >> 16) | (tile_x << 16));  // tile_dim0
  g1[4] = (int)(tile_y & 0xffffu);              // tile_dim1 (tile_dim2=0)
  g1[5] = (int)stride;                          // tensor_dim0_stride[31:0]
  g1[6] = 0;                                    // stride[47:32], dim1_stride
  g1[7] = 0;
  const i32x4 gz = {0, 0, 0, 0};
#if TDM_ARGS6
  const i32x8 gz8 = {0, 0, 0, 0, 0, 0, 0, 0};
  __builtin_amdgcn_tensor_load_to_lds(g0, g1, gz, gz, gz8, 0);
#else
  __builtin_amdgcn_tensor_load_to_lds(g0, g1, gz, gz, 0);
#endif
}

// ---------------------------------------------------------------------------
// 1) GroupNorm: one block per (batch, group); float4 loads, f16 output.
// ---------------------------------------------------------------------------
__global__ __launch_bounds__(256) void gn_kernel(
    const float* __restrict__ x, const float* __restrict__ gamma,
    const float* __restrict__ beta, _Float16* __restrict__ h) {
  __shared__ float s_sum[256];
  __shared__ float s_sq[256];
  const int bg = blockIdx.x;
  const int b = bg >> 5, g = bg & 31;
  const float4* xp4 = (const float4*)(x + (long)(b * C_ + g * CPG_) * N_);
  float sum = 0.f, sq = 0.f;
  for (int c = threadIdx.x; c < CPG_ * N_ / 4; c += 256) {
    const float4 v = xp4[c];
    sum += v.x + v.y + v.z + v.w;
    sq += v.x * v.x + v.y * v.y + v.z * v.z + v.w * v.w;
  }
  s_sum[threadIdx.x] = sum;
  s_sq[threadIdx.x] = sq;
  __syncthreads();
  for (int off = 128; off > 0; off >>= 1) {
    if ((int)threadIdx.x < off) {
      s_sum[threadIdx.x] += s_sum[threadIdx.x + off];
      s_sq[threadIdx.x] += s_sq[threadIdx.x + off];
    }
    __syncthreads();
  }
  const float inv_n = 1.f / (float)(CPG_ * N_);
  const float mean = s_sum[0] * inv_n;
  const float var = s_sq[0] * inv_n - mean * mean;
  const float rinv = rsqrtf(var + 1e-5f);
  _Float16* hp = h + (long)(b * C_ + g * CPG_) * N_;
  for (int c = threadIdx.x; c < CPG_ * N_ / 4; c += 256) {
    const int ch = g * CPG_ + (c >> 8);
    const float gm = gamma[ch] * rinv, bt = beta[ch];
    const float4 v = xp4[c];
    v4h o;
    o[0] = (_Float16)((v.x - mean) * gm + bt);
    o[1] = (_Float16)((v.y - mean) * gm + bt);
    o[2] = (_Float16)((v.z - mean) * gm + bt);
    o[3] = (_Float16)((v.w - mean) * gm + bt);
    *(v4h*)(hp + c * 4) = o;
  }
}

// ---------------------------------------------------------------------------
// 2) Convert GEMM weights to f16 (float4 -> v4h).
// ---------------------------------------------------------------------------
__global__ __launch_bounds__(256) void cvt_kernel(
    const float* __restrict__ qkvw, const float* __restrict__ projw,
    _Float16* __restrict__ qkvw_h, _Float16* __restrict__ projw_h) {
  const int i = blockIdx.x * 256 + threadIdx.x;
  if (i < 3 * C_ * C_ / 4) {
    const float4 v = ((const float4*)qkvw)[i];
    v4h o;
    o[0] = (_Float16)v.x; o[1] = (_Float16)v.y;
    o[2] = (_Float16)v.z; o[3] = (_Float16)v.w;
    ((v4h*)qkvw_h)[i] = o;
  }
  if (i < C_ * C_ / 4) {
    const float4 v = ((const float4*)projw)[i];
    v4h o;
    o[0] = (_Float16)v.x; o[1] = (_Float16)v.y;
    o[2] = (_Float16)v.z; o[3] = (_Float16)v.w;
    ((v4h*)projw_h)[i] = o;
  }
}

// ---------------------------------------------------------------------------
// 3) QKV GEMM: qkv[b,o,n] = sum_c W[o,c]*h[b,c,n] + bias[o].
// ---------------------------------------------------------------------------
__global__ __launch_bounds__(256) void qkv_gemm(
    const _Float16* __restrict__ w, const float* __restrict__ bias,
    const _Float16* __restrict__ h, _Float16* __restrict__ qkv) {
  __shared__ __align__(16) _Float16 Bt[64][32];  // [n][k] transposed tile
  const int wave = threadIdx.x >> 5;
  const int lane = threadIdx.x & 31;
  const int osub = wave >> 2, nsub = wave & 3;
  const int nt = blockIdx.x;
  const int ot = blockIdx.y * 2 + osub;
  const int b = blockIdx.z;
  const _Float16* hb = h + (long)b * C_ * N_ + nt * 64;
  const _Float16* wb = w + (long)ot * 16 * C_;
  const int cst = threadIdx.x >> 3;
  const int n8 = (threadIdx.x & 7) * 8;
  v8f acc = {};
  for (int kk = 0; kk < C_; kk += 32) {
    __syncthreads();
    {
      const v8h val = *(const v8h*)(hb + (long)(kk + cst) * N_ + n8);
#pragma unroll
      for (int i = 0; i < 8; ++i) Bt[n8 + i][cst] = val[i];
      if (kk + 32 < C_)
        __builtin_prefetch(hb + (long)(kk + 32 + cst) * N_ + n8, 0, 3);
    }
    __syncthreads();
    const v16h a = frag_a(wb + kk, C_);
    const v16h bf = frag_a(&Bt[nsub * 16][0], 32);
    acc = wmma_f16(a, bf, acc);
  }
  const int ncol = lane & 15, hh = lane >> 4;
#pragma unroll
  for (int v = 0; v < 8; ++v) {
    const int m = v + 8 * hh;
    const int o = ot * 16 + m;
    qkv[((long)b * 768 + o) * N_ + nt * 64 + nsub * 16 + ncol] =
        (_Float16)(acc[v] + bias[o]);
  }
}

// ---------------------------------------------------------------------------
// 4) Flash attention: block = 4 waves = 64 queries of one (b, head).
//    K staged transposed by VALU; V tile DMA'd by the TDM into LDS.
// ---------------------------------------------------------------------------
__global__ __launch_bounds__(128) void attn_kernel(
    const _Float16* __restrict__ qkv, _Float16* __restrict__ out) {
  __shared__ __align__(16) _Float16 Qt[64][64];     // [qrow][d]
  __shared__ __align__(16) _Float16 Kt[32][64];     // [key][d]
  __shared__ __align__(16) _Float16 Vt[64][32];     // [d][key] (natural)
  __shared__ __align__(16) _Float16 Pl[4][16][32];  // per-wave P staging

  const int wave = threadIdx.x >> 5;
  const int lane = threadIdx.x & 31;
  const int qt = blockIdx.x;
  const int hd = blockIdx.y;
  const int b = blockIdx.z;

  const long base = ((long)b * 768 + hd * 192) * N_;
  const _Float16* qp = qkv + base;
  const _Float16* kp = qkv + base + (long)64 * N_;
  const _Float16* vp = qkv + base + (long)128 * N_;
  const unsigned vt_lds = (unsigned)(uintptr_t)&Vt[0][0];

  // Stage Q transposed: 64x64, wide global reads, scatter to LDS.
  for (int c = threadIdx.x; c < 512; c += 128) {
    const int dd = c >> 3, q8 = (c & 7) * 8;
    const v8h val = *(const v8h*)(qp + (long)dd * N_ + qt * 64 + q8);
#pragma unroll
    for (int i = 0; i < 8; ++i) Qt[q8 + i][dd] = val[i];
  }
  __syncthreads();
  const v16h qa0 = frag_a(&Qt[wave * 16][0], 64);
  const v16h qa1 = frag_a(&Qt[wave * 16][32], 64);

  const float scale = 0.125f;  // 64^-0.5
  float mrun[8], lrun[8];
  v8f acc[4] = {};
#pragma unroll
  for (int v = 0; v < 8; ++v) {
    mrun[v] = -1e30f;
    lrun[v] = 0.f;
  }

  for (int kb = 0; kb < N_; kb += 32) {
    __syncthreads();
    // Async DMA: V tile [64 rows d][32 keys], row stride N -> compact LDS.
    if (wave == 0) tdm_load_tile_f16(vt_lds, vp + kb, 32, 64, N_);
    // Meanwhile stage K transposed with VALU/DS.
    for (int c = threadIdx.x; c < 256; c += 128) {
      const int dd = c >> 2, j8 = (c & 3) * 8;
      const v8h kv = *(const v8h*)(kp + (long)dd * N_ + kb + j8);
#pragma unroll
      for (int i = 0; i < 8; ++i) Kt[j8 + i][dd] = kv[i];
      if (kb + 32 < N_)
        __builtin_prefetch(kp + (long)dd * N_ + kb + 32 + j8, 0, 3);
    }
    if (wave == 0) __builtin_amdgcn_s_wait_tensorcnt(0);
    __syncthreads();

    v8f s0 = {}, s1 = {};
    s0 = wmma_f16(qa0, frag_a(&Kt[0][0], 64), s0);
    s0 = wmma_f16(qa1, frag_a(&Kt[0][32], 64), s0);
    s1 = wmma_f16(qa0, frag_a(&Kt[16][0], 64), s1);
    s1 = wmma_f16(qa1, frag_a(&Kt[16][32], 64), s1);

    float p0[8], p1[8];
#pragma unroll
    for (int v = 0; v < 8; ++v) {
      const float a0 = s0[v] * scale;
      const float a1 = s1[v] * scale;
      float mx = fmaxf(a0, a1);
#pragma unroll
      for (int off = 1; off < 16; off <<= 1)
        mx = fmaxf(mx, __shfl_xor(mx, off, 32));
      const float mnew = fmaxf(mrun[v], mx);
      const float alpha = __expf(mrun[v] - mnew);
      const float e0 = __expf(a0 - mnew);
      const float e1 = __expf(a1 - mnew);
      float rs = e0 + e1;
#pragma unroll
      for (int off = 1; off < 16; off <<= 1) rs += __shfl_xor(rs, off, 32);
      lrun[v] = lrun[v] * alpha + rs;
      mrun[v] = mnew;
      p0[v] = e0;
      p1[v] = e1;
#pragma unroll
      for (int t = 0; t < 4; ++t) acc[t][v] *= alpha;
    }

    {
      const int ncol = lane & 15, hh = lane >> 4;
#pragma unroll
      for (int v = 0; v < 8; ++v) {
        const int m = v + 8 * hh;
        Pl[wave][m][ncol] = (_Float16)p0[v];
        Pl[wave][m][ncol + 16] = (_Float16)p1[v];
      }
    }
    __builtin_amdgcn_wave_barrier();

    const v16h pa = frag_a(&Pl[wave][0][0], 32);
#pragma unroll
    for (int t = 0; t < 4; ++t) {
      const v16h vb = frag_a(&Vt[t * 16][0], 32);
      acc[t] = wmma_f16(pa, vb, acc[t]);
    }
  }

  const int ncol = lane & 15, hh = lane >> 4;
#pragma unroll
  for (int v = 0; v < 8; ++v) {
    const float inv = 1.f / lrun[v];
    const int m = v + 8 * hh;
    const int n = qt * 64 + wave * 16 + m;
#pragma unroll
    for (int t = 0; t < 4; ++t) {
      const int cc = hd * 64 + t * 16 + ncol;
      out[((long)b * C_ + cc) * N_ + n] = (_Float16)(acc[t][v] * inv);
    }
  }
}

// ---------------------------------------------------------------------------
// 5) Proj GEMM + bias + residual, fp32 output.
// ---------------------------------------------------------------------------
__global__ __launch_bounds__(256) void proj_gemm(
    const _Float16* __restrict__ w, const float* __restrict__ bias,
    const _Float16* __restrict__ a_in, const float* __restrict__ x,
    float* __restrict__ out) {
  __shared__ __align__(16) _Float16 Bt[64][32];
  const int wave = threadIdx.x >> 5;
  const int lane = threadIdx.x & 31;
  const int osub = wave >> 2, nsub = wave & 3;
  const int nt = blockIdx.x;
  const int ot = blockIdx.y * 2 + osub;
  const int b = blockIdx.z;
  const _Float16* ab = a_in + (long)b * C_ * N_ + nt * 64;
  const _Float16* wb = w + (long)ot * 16 * C_;
  const int cst = threadIdx.x >> 3;
  const int n8 = (threadIdx.x & 7) * 8;
  v8f acc = {};
  for (int kk = 0; kk < C_; kk += 32) {
    __syncthreads();
    {
      const v8h val = *(const v8h*)(ab + (long)(kk + cst) * N_ + n8);
#pragma unroll
      for (int i = 0; i < 8; ++i) Bt[n8 + i][cst] = val[i];
      if (kk + 32 < C_)
        __builtin_prefetch(ab + (long)(kk + 32 + cst) * N_ + n8, 0, 3);
    }
    __syncthreads();
    const v16h af = frag_a(wb + kk, C_);
    const v16h bf = frag_a(&Bt[nsub * 16][0], 32);
    acc = wmma_f16(af, bf, acc);
  }
  const int ncol = lane & 15, hh = lane >> 4;
#pragma unroll
  for (int v = 0; v < 8; ++v) {
    const int m = v + 8 * hh;
    const int o = ot * 16 + m;
    const long idx = ((long)b * C_ + o) * N_ + nt * 64 + nsub * 16 + ncol;
    out[idx] = acc[v] + bias[o] + x[idx];
  }
}

// ---------------------------------------------------------------------------
extern "C" void kernel_launch(void* const* d_in, const int* in_sizes, int n_in,
                              void* d_out, int out_size, void* d_ws,
                              size_t ws_size, hipStream_t stream) {
  (void)in_sizes; (void)n_in; (void)out_size; (void)ws_size;
  const float* x = (const float*)d_in[0];
  const float* gamma = (const float*)d_in[1];
  const float* beta = (const float*)d_in[2];
  const float* qkvw = (const float*)d_in[3];
  const float* qkvb = (const float*)d_in[4];
  const float* projw = (const float*)d_in[5];
  const float* projb = (const float*)d_in[6];
  float* out = (float*)d_out;

  char* ws = (char*)d_ws;
  _Float16* h      = (_Float16*)(ws);                              //  8 MiB
  _Float16* qkv    = (_Float16*)(ws + (8l << 20));                 // 24 MiB
  _Float16* attno  = (_Float16*)(ws + (32l << 20));                //  8 MiB
  _Float16* qkvwh  = (_Float16*)(ws + (40l << 20));                // 384 KiB
  _Float16* projwh = (_Float16*)(ws + (40l << 20) + (512l << 10)); // 128 KiB

  gn_kernel<<<dim3(B_ * GROUPS_), 256, 0, stream>>>(x, gamma, beta, h);
  cvt_kernel<<<dim3(3 * C_ * C_ / 4 / 256), 256, 0, stream>>>(qkvw, projw,
                                                              qkvwh, projwh);
  qkv_gemm<<<dim3(16, 24, B_), 256, 0, stream>>>(qkvwh, qkvb, h, qkv);
  attn_kernel<<<dim3(16, HEADS_, B_), 128, 0, stream>>>(qkv, attno);
  proj_gemm<<<dim3(16, 8, B_), 256, 0, stream>>>(projwh, projb, attno, x, out);
}
